// GINModel_23665269801080
// MI455X (gfx1250) — compile-verified, matrix operand order checked by reading, not run on hardware
//
#include <hip/hip_runtime.h>
#include <hip/hip_bf16.h>

#define NN 100000          // nodes
#define NE 1600000         // edges
#define NG 512             // graphs
#define DIM 128            // embedding dim
#define HID 256            // hidden dim
#define NL 5               // layers
#define BN_EPS 1e-5f
#define MTILE 128          // rows per MLP block

typedef __bf16 bf16;
typedef __attribute__((ext_vector_type(16))) __bf16 v16bf;
typedef __attribute__((ext_vector_type(8)))  float  v8f;
typedef __attribute__((ext_vector_type(4)))  unsigned int v4u;
typedef __attribute__((ext_vector_type(8)))  int v8i;
typedef __attribute__((ext_vector_type(4)))  int v4i;

// LDS layout offsets for mlp_kernel (dynamic LDS assumed to start at 0)
#define LDS_ZT   0u                       // [MTILE][DIM] bf16   32 KB
#define LDS_HID  32768u                   // [MTILE][HID] bf16   64 KB
#define LDS_W1   (32768u + 65536u)        // [HID][DIM]  bf16    64 KB
#define LDS_W2   (32768u + 2u * 65536u)   // [DIM][HID]  bf16    64 KB
#define LDS_BNL  (32768u + 3u * 65536u)   // [2][DIM]  f32        1 KB
#define LDS_B1   (LDS_BNL + 1024u)        // [HID] f32            1 KB
#define LDS_B2   (LDS_B1 + 1024u)         // [DIM] f32          0.5 KB
#define LDS_TOTAL (LDS_B2 + 512u)         // 231936 B

// ---------------------------------------------------------------------------
// Fragment loader for V_WMMA_F32_16X16X32_BF16 (A, or B via [N][K] storage).
// Lanes 0-15: row=lane, K=kb+0..7 (VGPR0-3) and kb+16..23 (VGPR4-7);
// lanes 16-31: K offset +8 (caller folds via kb = kk + (lane>>4)*8).
// Contiguous-K storage, 16B aligned -> two ds_load_b128.
// ---------------------------------------------------------------------------
__device__ inline v16bf load_frag(const bf16* base, int stride, int r, int kb) {
    union { v16bf v; uint4 q[2]; } f;
    const char* p = (const char*)(base + r * stride + kb);
    f.q[0] = *(const uint4*)(p);
    f.q[1] = *(const uint4*)(p + 32);
    return f.v;
}

// ---------------------------------------------------------------------------
// Tensor Data Mover: 2D bf16 tile, global -> LDS (D# per ISA 08 / sec 8).
// group0: count=1 | lds_addr | global_addr[56:0] | type=2
// group1: data_size=2B, tensor dim0/dim1, tile == tensor, dim0 stride
// ---------------------------------------------------------------------------
__device__ inline void tdm_load_2d_bf16(const void* gptr, unsigned lds_off,
                                        unsigned dim0, unsigned dim1,
                                        unsigned stride0) {
    unsigned long long ga = (unsigned long long)gptr;
    v4u g0;
    g0[0] = 1u;                                   // count=1, user mode
    g0[1] = lds_off;                              // lds_addr (bytes)
    g0[2] = (unsigned)(ga & 0xFFFFFFFFu);         // global_addr lo
    g0[3] = (unsigned)((ga >> 32) & 0x01FFFFFFu) | (2u << 30); // hi | type=2
    v8i g1;
    g1[0] = (int)(1u << 16);                      // data_size=1 (2 bytes)
    g1[1] = (int)((dim0 & 0xFFFFu) << 16);        // tensor_dim0[15:0]
    g1[2] = (int)((dim0 >> 16) | ((dim1 & 0xFFFFu) << 16));
    g1[3] = (int)((dim1 >> 16) | ((dim0 & 0xFFFFu) << 16)); // tile_dim0=dim0
    g1[4] = (int)(dim1 & 0xFFFFu);                // tile_dim1=dim1, tile_dim2=0
    g1[5] = (int)stride0;                         // tensor_dim0_stride lo
    g1[6] = 0;
    g1[7] = 0;
    v4i z4 = {0, 0, 0, 0};
#if __clang_major__ >= 23
    v8i z8 = {0, 0, 0, 0, 0, 0, 0, 0};
    __builtin_amdgcn_tensor_load_to_lds(g0, g1, z4, z4, z8, 0);
#else
    __builtin_amdgcn_tensor_load_to_lds(g0, g1, z4, z4, 0);
#endif
}

// ---------------------------------------------------------------------------
// h[i] = node_emb[x[i]]
// ---------------------------------------------------------------------------
__global__ void embed_kernel(const int* __restrict__ x,
                             const float* __restrict__ node_emb,
                             float* __restrict__ h) {
    int i = blockIdx.x * blockDim.x + threadIdx.x;
    if (i >= NN * DIM) return;
    int node = i >> 7, f = i & (DIM - 1);
    h[i] = node_emb[x[node] * DIM + f];
}

__global__ void copy4_kernel(const float4* __restrict__ src, float4* __restrict__ dst, int n4) {
    int i = blockIdx.x * blockDim.x + threadIdx.x;
    if (i < n4) dst[i] = src[i];
}

__global__ void zero_kernel(float* __restrict__ p, int n) {
    int i = blockIdx.x * blockDim.x + threadIdx.x;
    if (i < n) p[i] = 0.0f;
}

// ---------------------------------------------------------------------------
// Message + scatter: one wave32 per edge, one float4 per lane.
// z[dst] += relu(h[src] + edge_emb[attr])   (z was pre-initialized to h)
// ---------------------------------------------------------------------------
__global__ __launch_bounds__(256)
void msg_kernel(const float* __restrict__ h,
                const int* __restrict__ eidx,       // [2][NE]
                const int* __restrict__ eattr,      // [NE]
                const float* __restrict__ edge_emb, // [5][DIM]
                float* __restrict__ z) {
    __shared__ float eemb[5 * DIM];
    for (int i = threadIdx.x; i < 5 * DIM; i += blockDim.x) eemb[i] = edge_emb[i];
    __syncthreads();

    int gt = blockIdx.x * blockDim.x + threadIdx.x;
    int e = gt >> 5;
    int lane = gt & 31;
    if (e >= NE) return;

    if (lane == 0 && e + 2048 < NE) {              // global_prefetch_b8
        __builtin_prefetch(&eidx[e + 2048], 0, 0);
        __builtin_prefetch(&eidx[NE + e + 2048], 0, 0);
    }

    int src = eidx[e];
    int dst = eidx[NE + e];
    int a   = eattr[e];

    const float4 hv = *(const float4*)(h + (size_t)src * DIM + lane * 4);
    const float4 ev = *(const float4*)(eemb + a * DIM + lane * 4);
    float4 m;
    m.x = fmaxf(hv.x + ev.x, 0.0f);
    m.y = fmaxf(hv.y + ev.y, 0.0f);
    m.z = fmaxf(hv.z + ev.z, 0.0f);
    m.w = fmaxf(hv.w + ev.w, 0.0f);

    float* zp = z + (size_t)dst * DIM + lane * 4;
    atomicAdd(zp + 0, m.x);
    atomicAdd(zp + 1, m.y);
    atomicAdd(zp + 2, m.z);
    atomicAdd(zp + 3, m.w);
}

// ---------------------------------------------------------------------------
// Per-layer weight conversion: fp32 -> bf16 transposed to [N][K].
// ---------------------------------------------------------------------------
__global__ void convert_weights(const float* __restrict__ W1l,
                                const float* __restrict__ W2l,
                                bf16* __restrict__ w1t,
                                bf16* __restrict__ w2t) {
    int i = blockIdx.x * blockDim.x + threadIdx.x;
    if (i >= DIM * HID) return;
    {
        int n = i / DIM, k = i % DIM;
        w1t[i] = (bf16)W1l[k * HID + n];
    }
    {
        int n = i / HID, k = i % HID;
        w2t[i] = (bf16)W2l[k * DIM + n];
    }
}

// ---------------------------------------------------------------------------
// Fused MLP: zout = relu(z @ W1 + b1) @ W2 + b2 on a 128-row tile, plus BN
// partial sums. Weights staged by the Tensor Data Mover (tensor_load_to_lds,
// TENSORcnt), activations by vector loads; both GEMMs on WMMA bf16.
// ---------------------------------------------------------------------------
__global__ __launch_bounds__(256)
void mlp_kernel(const float* __restrict__ z,
                const bf16* __restrict__ w1t, const bf16* __restrict__ w2t,
                const float* __restrict__ b1, const float* __restrict__ b2,
                float* __restrict__ zout,
                float* __restrict__ bnsum, float* __restrict__ bnsumsq) {
    extern __shared__ char smem[];
    bf16*  zt  = (bf16*)(smem + LDS_ZT);
    bf16*  hid = (bf16*)(smem + LDS_HID);
    bf16*  w1s = (bf16*)(smem + LDS_W1);
    bf16*  w2s = (bf16*)(smem + LDS_W2);
    float* bnl = (float*)(smem + LDS_BNL);
    float* b1s = (float*)(smem + LDS_B1);
    float* b2s = (float*)(smem + LDS_B2);

    const int tid  = threadIdx.x;
    const int wave = tid >> 5;
    const int lane = tid & 31;
    const int lh   = lane >> 4;   // wave half (K offset +8)
    const int ll   = lane & 15;
    const int row0 = blockIdx.x * MTILE;

    // Wave 0: DMA both weight matrices into LDS via the TDM, then drain
    // TENSORcnt before the workgroup barrier publishes them.
    if (wave == 0) {
        tdm_load_2d_bf16(w1t, LDS_W1, DIM, HID, DIM);   // [HID][DIM] bf16
        tdm_load_2d_bf16(w2t, LDS_W2, HID, DIM, HID);   // [DIM][HID] bf16
        __builtin_amdgcn_s_wait_tensorcnt(0);
    }

    // Stage activation tile fp32 -> bf16 (OOB rows zero-filled)
    for (int i = tid; i < MTILE * DIM; i += 256) {
        int r = i >> 7, c = i & (DIM - 1);
        int gr = row0 + r;
        float v = (gr < NN) ? z[(size_t)gr * DIM + c] : 0.0f;
        zt[i] = (bf16)v;
    }
    // Stage biases + zero BN scratch
    for (int i = tid; i < HID; i += 256) b1s[i] = b1[i];
    for (int i = tid; i < DIM; i += 256) b2s[i] = b2[i];
    for (int i = tid; i < 2 * DIM; i += 256) bnl[i] = 0.0f;
    __syncthreads();

    // Phase 1: hid = relu(zt @ W1 + b1); 8x16 tiles of 16x16 over 8 waves.
    // All 4 k-step fragments preloaded so the DS loads pipeline and the
    // 4 WMMAs issue back-to-back.
    for (int t = wave; t < (MTILE / 16) * (HID / 16); t += 8) {
        int mt = t >> 4;          // 0..7
        int nt = t & 15;          // 0..15
        v16bf fa[4], fb[4];
#pragma unroll
        for (int s = 0; s < 4; ++s) {
            int kb = s * 32 + lh * 8;
            fa[s] = load_frag(zt,  DIM, mt * 16 + ll, kb);
            fb[s] = load_frag(w1s, DIM, nt * 16 + ll, kb);
        }
        v8f acc = {};
#pragma unroll
        for (int s = 0; s < 4; ++s)
            acc = __builtin_amdgcn_wmma_f32_16x16x32_bf16(
                false, fa[s], false, fb[s], (short)0, acc, false, false);
        int col = nt * 16 + ll;
        float bias = b1s[col];
#pragma unroll
        for (int v = 0; v < 8; ++v) {
            int r = mt * 16 + v + lh * 8;   // C/D layout: VGPR v -> row v (+8 hi half)
            hid[r * HID + col] = (bf16)fmaxf(acc[v] + bias, 0.0f);
        }
    }
    __syncthreads();

    // Phase 2: out = hid @ W2 + b2; 8x8 tiles over 8 waves; K=256 in two
    // preloaded groups of 4 steps. BN stats fused into the epilogue.
    for (int t = wave; t < (MTILE / 16) * (DIM / 16); t += 8) {
        int mt = t >> 3;          // 0..7
        int nt = t & 7;           // 0..7
        v8f acc = {};
#pragma unroll
        for (int g = 0; g < 2; ++g) {
            v16bf fa[4], fb[4];
#pragma unroll
            for (int s = 0; s < 4; ++s) {
                int kb = g * 128 + s * 32 + lh * 8;
                fa[s] = load_frag(hid, HID, mt * 16 + ll, kb);
                fb[s] = load_frag(w2s, HID, nt * 16 + ll, kb);
            }
#pragma unroll
            for (int s = 0; s < 4; ++s)
                acc = __builtin_amdgcn_wmma_f32_16x16x32_bf16(
                    false, fa[s], false, fb[s], (short)0, acc, false, false);
        }
        int col = nt * 16 + ll;
        float bias = b2s[col];
        float s = 0.0f, sq = 0.0f;
#pragma unroll
        for (int v = 0; v < 8; ++v) {
            int r  = mt * 16 + v + lh * 8;
            int gr = row0 + r;
            if (gr < NN) {
                float val = acc[v] + bias;
                zout[(size_t)gr * DIM + col] = val;
                s  += val;
                sq += val * val;
            }
        }
        atomicAdd(&bnl[col], s);            // ds_add_f32
        atomicAdd(&bnl[DIM + col], sq);
    }
    __syncthreads();

    for (int i = tid; i < DIM; i += 256) {
        atomicAdd(&bnsum[i],   bnl[i]);
        atomicAdd(&bnsumsq[i], bnl[DIM + i]);
    }
}

// ---------------------------------------------------------------------------
// BatchNorm finalize + relu:  h = relu(gamma*(z-mu)*rsqrt(var+eps)+beta)
// ---------------------------------------------------------------------------
__global__ void bn_relu_kernel(const float* __restrict__ z,
                               const float* __restrict__ bnsum,
                               const float* __restrict__ bnsumsq,
                               const float* __restrict__ gamma,
                               const float* __restrict__ beta,
                               float* __restrict__ h) {
    int i = blockIdx.x * blockDim.x + threadIdx.x;
    if (i >= NN * DIM) return;
    int f = i & (DIM - 1);
    const float inv = 1.0f / (float)NN;
    float mu  = bnsum[f] * inv;
    float var = bnsumsq[f] * inv - mu * mu;
    float val = gamma[f] * (z[i] - mu) * rsqrtf(var + BN_EPS) + beta[f];
    h[i] = fmaxf(val, 0.0f);
}

// ---------------------------------------------------------------------------
// global_mean_pool
// ---------------------------------------------------------------------------
__global__ void pool_acc_kernel(const float* __restrict__ h,
                                const int* __restrict__ batch,
                                float* __restrict__ psum,
                                float* __restrict__ pcnt) {
    int i = blockIdx.x * blockDim.x + threadIdx.x;
    if (i >= NN * DIM) return;
    int node = i >> 7, f = i & (DIM - 1);
    int g = batch[node];
    atomicAdd(&psum[g * DIM + f], h[i]);
    if (f == 0) atomicAdd(&pcnt[g], 1.0f);
}

__global__ void pool_div_kernel(const float* __restrict__ psum,
                                const float* __restrict__ pcnt,
                                float* __restrict__ out) {
    int i = blockIdx.x * blockDim.x + threadIdx.x;
    if (i >= NG * DIM) return;
    int g = i >> 7;
    out[i] = psum[i] / fmaxf(pcnt[g], 1.0f);
}

// ---------------------------------------------------------------------------
extern "C" void kernel_launch(void* const* d_in, const int* in_sizes, int n_in,
                              void* d_out, int out_size, void* d_ws, size_t ws_size,
                              hipStream_t stream) {
    const int*   x        = (const int*)d_in[0];
    const int*   eidx     = (const int*)d_in[1];
    const int*   eattr    = (const int*)d_in[2];
    const int*   batch    = (const int*)d_in[3];
    const float* node_emb = (const float*)d_in[4];
    const float* edge_emb = (const float*)d_in[5];
    const float* W1       = (const float*)d_in[6];
    const float* b1       = (const float*)d_in[7];
    const float* W2       = (const float*)d_in[8];
    const float* b2       = (const float*)d_in[9];
    const float* gammaA   = (const float*)d_in[10];
    const float* betaA    = (const float*)d_in[11];
    float*       out      = (float*)d_out;

    char* ws = (char*)d_ws;
    const size_t ND_BYTES = (size_t)NN * DIM * sizeof(float);   // 51.2 MB
    float* hbuf = (float*)(ws);
    float* zbuf = (float*)(ws + ND_BYTES);
    bf16*  w1t  = (bf16*)(ws + 2 * ND_BYTES);
    bf16*  w2t  = (bf16*)(ws + 2 * ND_BYTES + (size_t)DIM * HID * 2);
    float* bns  = (float*)(ws + 2 * ND_BYTES + 2 * (size_t)DIM * HID * 2);
    float* psum = bns + 2 * DIM;
    float* pcnt = psum + (size_t)NG * DIM;

    const int T = 256;
    const int gND  = (NN * DIM + T - 1) / T;
    const int gCPY = (NN * DIM / 4 + T - 1) / T;
    const int gMSG = (NE * 32 + T - 1) / T;
    const int gW   = (DIM * HID + T - 1) / T;
    const int gMLP = (NN + MTILE - 1) / MTILE;   // 782

    embed_kernel<<<gND, T, 0, stream>>>(x, node_emb, hbuf);

    for (int l = 0; l < NL; ++l) {
        convert_weights<<<gW, T, 0, stream>>>(W1 + (size_t)l * DIM * HID,
                                              W2 + (size_t)l * HID * DIM, w1t, w2t);
        copy4_kernel<<<gCPY, T, 0, stream>>>((const float4*)hbuf, (float4*)zbuf, NN * DIM / 4);
        msg_kernel<<<gMSG, T, 0, stream>>>(hbuf, eidx, eattr, edge_emb, zbuf);
        zero_kernel<<<1, T, 0, stream>>>(bns, 2 * DIM);
        mlp_kernel<<<gMLP, T, LDS_TOTAL, stream>>>(zbuf, w1t, w2t,
                                                   b1 + (size_t)l * HID,
                                                   b2 + (size_t)l * DIM,
                                                   zbuf, bns, bns + DIM);
        bn_relu_kernel<<<gND, T, 0, stream>>>(zbuf, bns, bns + DIM,
                                              gammaA + (size_t)l * DIM,
                                              betaA + (size_t)l * DIM, hbuf);
    }

    zero_kernel<<<(NG * DIM + NG + T - 1) / T, T, 0, stream>>>(psum, NG * DIM + NG);
    pool_acc_kernel<<<gND, T, 0, stream>>>(hbuf, batch, psum, pcnt);
    pool_div_kernel<<<(NG * DIM + T - 1) / T, T, 0, stream>>>(psum, pcnt, out);

    (void)in_sizes; (void)n_in; (void)out_size; (void)ws_size;
}
